// EPN_flax_17729624998573
// MI455X (gfx1250) — matrix operand order repaired and checked by reading, not compiled, not run on hardware
//
#include <hip/hip_runtime.h>

// EPN (electron-passing network) for MI455X / gfx1250, wave32 + WMMA.
//
// Per timestep t:
//   a   = [h, q]  (N x 32)
//   P'  = a @ W0[t][0:32]  + b0  (N x 64)   -- b0 folded (appears once per direction)
//   Qm  = a @ W0[t][32:64]       (N x 64)
//   per pair (i,j):
//     E1^T   = W0e^T @ e^T        (WMMA, transposed so D lands in A-fragment lanes)
//     h1_ij  = relu(P'_i + Qm_j + E1),  h1_ji = relu(P'_j + Qm_i + E1)
//     h2_*   = relu(h1_* @ W1 + b1)     (WMMA 16x16x32, K=64)
//     antisym= (h2_ij - h2_ji) . W2 * mask      (b2 cancels in the difference)
//   q_i += sum_j antisym

#define NATOM 1024
#define DH    31
#define DE    16
#define HID   64
#define TSTEPS 3

typedef __attribute__((ext_vector_type(16))) _Float16 v16h;
typedef __attribute__((ext_vector_type(8)))  float    v8f;
typedef __attribute__((ext_vector_type(4)))  float    v4f;

// ---------------------------------------------------------------------------
// Prep: P' = a @ W0[t][0:32,:] + b0,  Qm = a @ W0[t][32:64,:]   (a = [h, q])
// ---------------------------------------------------------------------------
__global__ void prep_pq(const float* __restrict__ h, const float* __restrict__ qbuf,
                        const float* __restrict__ W0, const float* __restrict__ b0,
                        int t, float* __restrict__ P, float* __restrict__ Qm) {
  int idx = blockIdx.x * blockDim.x + threadIdx.x;   // 0 .. N*64-1
  int n = idx >> 6;
  int c = idx & 63;
  const float* w = W0 + (size_t)t * 80 * 64;
  float accP = 0.f, accQ = 0.f;
  #pragma unroll 1
  for (int k = 0; k < DH; ++k) {
    float av = h[n * DH + k];
    accP += av * w[k * 64 + c];
    accQ += av * w[(32 + k) * 64 + c];
  }
  float qv = qbuf[n];
  accP += qv * w[31 * 64 + c];
  accQ += qv * w[63 * 64 + c];
  P[n * 64 + c]  = accP + b0[t * 64 + c];
  Qm[n * 64 + c] = accQ;
}

// ---------------------------------------------------------------------------
// Pack W1 (f32 [t][64][64]) into per-lane B-fragment order, f16.
// B (32x16 f16, v16h per lane): lanes 0-15 -> col N=lane, K = kb*32 + 0..7, 16..23;
//                               lanes 16-31 -> K = kb*32 + 8..15, 24..31.
// Layout: W1p[t][kb][tile][lane][16]
// ---------------------------------------------------------------------------
__global__ void pack_w1(const float* __restrict__ W1, _Float16* __restrict__ W1p) {
  int idx = blockIdx.x * blockDim.x + threadIdx.x;   // total 3*2*4*32*16 = 12288
  if (idx >= TSTEPS * 2 * 4 * 32 * 16) return;
  int i16  = idx & 15;
  int lane = (idx >> 4) & 31;
  int tile = (idx >> 9) & 3;
  int kb   = (idx >> 11) & 1;
  int t    = idx >> 12;
  int half = lane >> 4, lm = lane & 15;
  int K    = kb * 32 + (i16 < 8 ? half * 8 + i16 : 16 + half * 8 + (i16 - 8));
  int Ncol = tile * 16 + lm;
  W1p[idx] = (_Float16)W1[((size_t)t * 64 + K) * 64 + Ncol];
}

// ---------------------------------------------------------------------------
// Pack W0e^T = W0[t][64:80,:]^T as A-fragments (M = hidden channel n-tile,
// K = e-dim 0..15 real, 16..31 zero).
// A (16x32 f16), contiguous K here: lane<16: M=lane, idx0..15 = K 0..15;
//                       lane>=16: M=lane-16, idx0..15 = K 16..31 (zero).
// Layout: W0eTp[t][tile][lane][16]
// ---------------------------------------------------------------------------
__global__ void pack_w0eT(const float* __restrict__ W0, _Float16* __restrict__ W0eTp) {
  int idx = blockIdx.x * blockDim.x + threadIdx.x;   // total 3*4*32*16 = 6144
  if (idx >= TSTEPS * 4 * 32 * 16) return;
  int i16  = idx & 15;
  int lane = (idx >> 4) & 31;
  int tile = (idx >> 9) & 3;
  int t    = idx >> 11;
  int half = lane >> 4, lm = lane & 15;
  _Float16 v = (_Float16)0.f;
  if (half == 0) {                     // lanes 16..31 -> K=16..31 -> zero pad
    int k = i16;                       // e-dimension (K)
    int n = tile * 16 + lm;            // hidden channel (M row of W0e^T)
    v = (_Float16)W0[((size_t)t * 80 + 64 + k) * 64 + n];
  }
  W0eTp[idx] = v;
}

__global__ void update_q(float* __restrict__ qbuf, const float* __restrict__ delta) {
  int i = blockIdx.x * blockDim.x + threadIdx.x;
  if (i < NATOM) qbuf[i] += delta[i];
}

// ---------------------------------------------------------------------------
// Main pairwise kernel. Block = 256 threads = 8 waves, no LDS.
// Wave w handles i = blockIdx.y*8 + w, pairs j in [j0, j0+16), j0 = blockIdx.x*16.
// ---------------------------------------------------------------------------
__global__ __launch_bounds__(256) void epn_main(
    const float* __restrict__ e, const float* __restrict__ mask,
    const float* __restrict__ P, const float* __restrict__ Qm,
    const float* __restrict__ b1, const float* __restrict__ W2,
    const _Float16* __restrict__ W1p, const _Float16* __restrict__ W0eTp,
    int t, float* __restrict__ delta)
{
  const int tid  = threadIdx.x;
  const int w    = tid >> 5;
  const int lane = tid & 31;
  const int half = lane >> 4;       // 0: lanes 0-15, 1: lanes 16-31
  const int lm   = lane & 15;
  const int i    = blockIdx.y * 8 + w;
  const int j0   = blockIdx.x * 16;
  const int jme  = j0 + lm;         // the j-row this lane owns

  // ---- B fragment for E1^T: column p = lm holds e[i][jme][0:16] ----
  // B 32x16 layout: lanes 0-15 -> K = 0..15 (real e dims); lanes 16-31 -> K = 16..31 (zero pad)
  const float* ep = e + (((size_t)i << 10) + (size_t)jme) * DE;
  v4f ev0 = *(const v4f*)(ep);
  v4f ev1 = *(const v4f*)(ep + 4);
  v4f ev2 = *(const v4f*)(ep + 8);
  v4f ev3 = *(const v4f*)(ep + 12);
  v16h bE;
  #pragma unroll
  for (int k = 0; k < 4; ++k) {
    bE[k]      = half ? (_Float16)0.f : (_Float16)ev0[k];
    bE[4 + k]  = half ? (_Float16)0.f : (_Float16)ev1[k];
    bE[8 + k]  = half ? (_Float16)0.f : (_Float16)ev2[k];
    bE[12 + k] = half ? (_Float16)0.f : (_Float16)ev3[k];
  }

  // ---- E1^T = W0e^T @ e^T : lane holds pair p=lm, channels n = tile*16 + 8*half + r ----
  const _Float16* w0base = W0eTp + (size_t)t * 4 * 512;
  v8f e1acc[4];
  #pragma unroll
  for (int tile = 0; tile < 4; ++tile) {
    v16h aF = *(const v16h*)(w0base + (size_t)tile * 512 + lane * 16);
    v8f c = {};
    e1acc[tile] = __builtin_amdgcn_wmma_f32_16x16x32_f16(
        false, aF, false, bE, (short)0, c, false, false);
  }

  // ---- Pre-sum both directions' layer-1 inputs (contiguous b128 loads) ----
  // s0 = E1 + P'_i + Qm_j  (direction ij),  s1 = E1 + P'_j + Qm_i  (direction ji)
  float s0[4][8], s1[4][8];
  #pragma unroll
  for (int tile = 0; tile < 4; ++tile) {
    int off = tile * 16 + half * 8;
    v4f piA = *(const v4f*)(P  + (size_t)i * 64 + off);
    v4f piB = *(const v4f*)(P  + (size_t)i * 64 + off + 4);
    v4f qiA = *(const v4f*)(Qm + (size_t)i * 64 + off);
    v4f qiB = *(const v4f*)(Qm + (size_t)i * 64 + off + 4);
    v4f pjA = *(const v4f*)(P  + (size_t)jme * 64 + off);
    v4f pjB = *(const v4f*)(P  + (size_t)jme * 64 + off + 4);
    v4f qjA = *(const v4f*)(Qm + (size_t)jme * 64 + off);
    v4f qjB = *(const v4f*)(Qm + (size_t)jme * 64 + off + 4);
    #pragma unroll
    for (int r = 0; r < 8; ++r) {
      float piv = (r < 4) ? piA[r & 3] : piB[r & 3];
      float qiv = (r < 4) ? qiA[r & 3] : qiB[r & 3];
      float pjv = (r < 4) ? pjA[r & 3] : pjB[r & 3];
      float qjv = (r < 4) ? qjA[r & 3] : qjB[r & 3];
      s0[tile][r] = e1acc[tile][r] + piv + qjv;
      s1[tile][r] = e1acc[tile][r] + pjv + qiv;
    }
  }

  // ---- Layer-2/3 broadcast operands (layer-2 D layout: row p = r+8*half, col n = lm+16*tile) ----
  float b1v[4], wm[4][8];
  float mk[8];
  #pragma unroll
  for (int r = 0; r < 8; ++r) mk[r] = mask[((size_t)i << 10) + j0 + r + half * 8];
  #pragma unroll
  for (int tile = 0; tile < 4; ++tile) {
    int n = tile * 16 + lm;
    b1v[tile] = b1[t * 64 + n];
    float w2v = W2[t * 64 + n];
    #pragma unroll
    for (int r = 0; r < 8; ++r) wm[tile][r] = w2v * mk[r];
  }

  const _Float16* w1base = W1p + (size_t)t * 4096;
  float part = 0.f;

  #pragma unroll
  for (int d = 0; d < 2; ++d) {
    // ---- A fragments built purely in-lane: transposed E1 puts channel subsets
    //      exactly where the 16x32 A layout wants them (bit3(K) == half) ----
    v16h aH[2];
    #pragma unroll
    for (int kb = 0; kb < 2; ++kb) {
      #pragma unroll
      for (int k = 0; k < 8; ++k) {
        float v0 = fmaxf(d ? s1[2 * kb][k]     : s0[2 * kb][k],     0.f);
        float v1 = fmaxf(d ? s1[2 * kb + 1][k] : s0[2 * kb + 1][k], 0.f);
        aH[kb][k]     = (_Float16)v0;
        aH[kb][8 + k] = (_Float16)v1;
      }
    }

    // ---- h2 = relu(h1 @ W1 + b1); fold layer-3 dot (.W2 * mask) ----
    float pd = 0.f;
    #pragma unroll
    for (int tile = 0; tile < 4; ++tile) {
      v8f acc;
      #pragma unroll
      for (int r = 0; r < 8; ++r) acc[r] = b1v[tile];
      #pragma unroll
      for (int kb = 0; kb < 2; ++kb) {
        v16h bF = *(const v16h*)(w1base + ((size_t)kb * 4 + tile) * 512 + lane * 16);
        acc = __builtin_amdgcn_wmma_f32_16x16x32_f16(
            false, aH[kb], false, bF, (short)0, acc, false, false);
      }
      #pragma unroll
      for (int r = 0; r < 8; ++r)
        pd += fmaxf(acc[r], 0.f) * wm[tile][r];
    }
    part += d ? -pd : pd;
  }

  // ---- Wave reduction: every (pair, channel) counted exactly once ----
  #pragma unroll
  for (int off = 16; off > 0; off >>= 1)
    part += __shfl_xor(part, off, 32);
  if (lane == 0) atomicAdd(&delta[i], part);
}

// ---------------------------------------------------------------------------
extern "C" void kernel_launch(void* const* d_in, const int* in_sizes, int n_in,
                              void* d_out, int out_size, void* d_ws, size_t ws_size,
                              hipStream_t stream) {
  (void)in_sizes; (void)n_in; (void)out_size; (void)ws_size;
  const float* h    = (const float*)d_in[0];
  const float* e    = (const float*)d_in[1];
  const float* q    = (const float*)d_in[2];
  const float* mask = (const float*)d_in[3];
  // d_in[4] = natom (compile-time constant here)
  const float* W0 = (const float*)d_in[5];
  const float* b0 = (const float*)d_in[6];
  const float* W1 = (const float*)d_in[7];
  const float* b1 = (const float*)d_in[8];
  const float* W2 = (const float*)d_in[9];
  // d_in[10] = b2: cancels in (elec_ij - elec_ji)
  float* out = (float*)d_out;

  char* ws = (char*)d_ws;
  float*    qbuf  = (float*)(ws);
  float*    delta = (float*)(ws + 4096);
  float*    P     = (float*)(ws + 8192);
  float*    Qm    = (float*)(ws + 8192 + 262144);
  _Float16* W1p   = (_Float16*)(ws + 8192 + 2 * 262144);
  _Float16* W0eTp = (_Float16*)(ws + 8192 + 2 * 262144 + 32768);

  hipMemcpyAsync(qbuf, q, NATOM * sizeof(float), hipMemcpyDeviceToDevice, stream);
  pack_w1<<<48, 256, 0, stream>>>(W1, W1p);
  pack_w0eT<<<24, 256, 0, stream>>>(W0, W0eTp);

  for (int t = 0; t < TSTEPS; ++t) {
    prep_pq<<<(NATOM * 64) / 256, 256, 0, stream>>>(h, qbuf, W0, b0, t, P, Qm);
    hipMemsetAsync(delta, 0, NATOM * sizeof(float), stream);
    epn_main<<<dim3(NATOM / 16, NATOM / 8), 256, 0, stream>>>(
        e, mask, P, Qm, b1, W2, W1p, W0eTp, t, delta);
    update_q<<<NATOM / 256, 256, 0, stream>>>(qbuf, delta);
  }
  hipMemcpyAsync(out, qbuf, NATOM * sizeof(float), hipMemcpyDeviceToDevice, stream);
}
